// HVTSurv_41901700940110
// MI455X (gfx1250) — matrix operand order repaired and checked by reading, not compiled, not run on hardware
//
#include <hip/hip_runtime.h>
#include <hip/hip_bf16.h>
#include <math.h>

typedef __attribute__((ext_vector_type(16))) _Float16 v16h;
typedef __attribute__((ext_vector_type(8)))  float    v8f;

#define DIM   512
#define NHEAD 8
#define WSZ   49
#define NWIN  1024
#define NTOK  (WSZ * NWIN)   /* 50176 */
#define QKSCALE 0.125f       /* (512/8)^-0.5 */

// ---------------------------------------------------------------------------
// CDNA5 async global->LDS copy (one b128 per lane), tracked by ASYNCcnt.
// LDS destination is a 32-bit byte address (flat LDS-aperture addr[31:0]).
// ---------------------------------------------------------------------------
__device__ __forceinline__ void async_copy_b128(_Float16* ldst,
                                                const _Float16* gsrc) {
  unsigned lds_addr = (unsigned)(uintptr_t)ldst;
  asm volatile("global_load_async_to_lds_b128 %0, %1, off"
               :
               : "v"(lds_addr), "v"(gsrc)
               : "memory");
}
__device__ __forceinline__ void wait_async0() {
  asm volatile("s_wait_asynccnt 0x0" ::: "memory");
}

// ---------------------------------------------------------------------------
// WMMA helpers (v_wmma_f32_16x16x32_f16, wave32)
// ---------------------------------------------------------------------------
__device__ __forceinline__ v8f wmma16(const v16h a, const v16h b, const v8f c) {
  return __builtin_amdgcn_wmma_f32_16x16x32_f16(false, a, false, b, (short)0, c,
                                                false, false);
}

// A fragment: 16x32 f16 per ISA layout (lanes 0-15: K {0-7,16-23};
// lanes 16-31: K {8-15,24-31}).
__device__ __forceinline__ v16h frag_a(const _Float16* base, int ld, int row,
                                       int khalf, int k0) {
  v16h a;
  const _Float16* p  = base + row * ld + k0 + khalf * 8;
  const _Float16* p2 = base + row * ld + k0 + 16 + khalf * 8;
#pragma unroll
  for (int e = 0; e < 8; ++e) { a[e] = p[e]; a[e + 8] = p2[e]; }
  return a;
}

// B fragment from B^T tile [N][K]: same K order as A (K {0-7,16-23} / {8-15,24-31}).
__device__ __forceinline__ v16h frag_b(const _Float16* base, int ld, int col,
                                       int khalf, int k0) {
  v16h b;
  const _Float16* p  = base + col * ld + k0 + khalf * 8;
  const _Float16* p2 = base + col * ld + k0 + 16 + khalf * 8;
#pragma unroll
  for (int e = 0; e < 8; ++e) { b[e] = p[e]; b[e + 8] = p2[e]; }
  return b;
}

template <int ACT>
__device__ __forceinline__ float apply_act(float v) {
  if (ACT == 1) return fmaxf(v, 0.0f);                                    // relu
  if (ACT == 2) return 0.5f * v * (1.0f + erff(v * 0.7071067811865476f)); // gelu
  if (ACT == 3) return tanhf(v);                                          // tanh
  return v;
}

// ---------------------------------------------------------------------------
// Weight pre-pass: WT[n][k] = (f16) W[k][n]   (W is K x N f32, row-major)
// ---------------------------------------------------------------------------
__global__ __launch_bounds__(256) void wt_kernel(const float* __restrict__ W,
                                                 _Float16* __restrict__ WT,
                                                 int K, int N) {
  long idx = (long)blockIdx.x * 256 + threadIdx.x;  // k*N + n, coalesced read
  if (idx < (long)K * N) {
    int k = (int)(idx / N);
    int n = (int)(idx % N);
    WT[(size_t)n * K + k] = (_Float16)W[idx];
  }
}

// ---------------------------------------------------------------------------
// Tile fills
// ---------------------------------------------------------------------------
template <bool A_F32>
__device__ __forceinline__ void fill_a(_Float16* sA, const void* Aptr, int lda,
                                       int mBlk, int k0, int tid) {
#pragma unroll
  for (int r = 0; r < 2; ++r) {           // 128 rows x 4 chunks = 512 / 256 thr
    int id = tid + r * 256;
    int mr = id >> 2;
    int c  = (id & 3) * 8;
    if (A_F32) {
      const float* src = (const float*)Aptr + (size_t)(mBlk + mr) * lda + k0 + c;
      float2 f0 = *(const float2*)(src);      // 8B-aligned (data+2 offset)
      float2 f1 = *(const float2*)(src + 2);
      float2 f2 = *(const float2*)(src + 4);
      float2 f3 = *(const float2*)(src + 6);
      union { _Float16 h[8]; uint4 u; } t;
      t.h[0] = (_Float16)f0.x; t.h[1] = (_Float16)f0.y;
      t.h[2] = (_Float16)f1.x; t.h[3] = (_Float16)f1.y;
      t.h[4] = (_Float16)f2.x; t.h[5] = (_Float16)f2.y;
      t.h[6] = (_Float16)f3.x; t.h[7] = (_Float16)f3.y;
      *(uint4*)&sA[mr * 32 + c] = t.u;
    } else {
      async_copy_b128(&sA[mr * 32 + c],
                      (const _Float16*)Aptr + (size_t)(mBlk + mr) * lda + k0 + c);
    }
  }
}

__device__ __forceinline__ void fill_b(_Float16* sBT, const _Float16* WT, int K,
                                       int nBlk, int k0, int tid) {
  int nr = tid >> 2;                      // 64 rows x 4 chunks = 256 / 256 thr
  int c  = (tid & 3) * 8;
  async_copy_b128(&sBT[nr * 32 + c], WT + (size_t)(nBlk + nr) * K + k0 + c);
}

// ---------------------------------------------------------------------------
// Tiled WMMA GEMM: Out[M,N] = act( A[M,K] @ W[K,N] + bias (+ resid) )
// 128x64 tile per block (8 waves, 2x2 WMMA subtiles per wave), ping-pong LDS,
// async global->LDS copies. Requires M%128==0, N%64==0, K%32==0.
// ---------------------------------------------------------------------------
template <bool A_F32, int ACT, bool RESID, bool OUT16>
__global__ __launch_bounds__(256) void gemm_kernel(
    const void* __restrict__ Aptr, int lda,
    const _Float16* __restrict__ WT,       // [N][K] f16 (pre-transposed)
    const float* __restrict__ bias,
    const float* __restrict__ resid,
    void* __restrict__ Out, int ldo,
    int M, int N, int K) {
  __shared__ _Float16 sA[2][128 * 32];
  __shared__ _Float16 sBT[2][64 * 32];
  const int tid   = threadIdx.x;
  const int lane  = tid & 31;
  const int wave  = tid >> 5;
  const int khalf = lane >> 4;
  const int r16   = lane & 15;
  const int mBlk  = blockIdx.y * 128;
  const int nBlk  = blockIdx.x * 64;
  const int mP    = (wave & 3) * 2;   // first of 2 m-subtiles
  const int nP    = (wave >> 2) * 2;  // first of 2 n-subtiles

  v8f acc[2][2] = {};

  fill_a<A_F32>(sA[0], Aptr, lda, mBlk, 0, tid);
  fill_b(sBT[0], WT, K, nBlk, 0, tid);

  int cur = 0;
  for (int k0 = 0; k0 < K; k0 += 32) {
    wait_async0();        // own async tile copies complete
    __syncthreads();      // all waves' copies complete
    if (k0 + 32 < K) {
      fill_a<A_F32>(sA[cur ^ 1], Aptr, lda, mBlk, k0 + 32, tid);
      fill_b(sBT[cur ^ 1], WT, K, nBlk, k0 + 32, tid);
    }
    const _Float16* a_ = sA[cur];
    const _Float16* b_ = sBT[cur];
    v16h af0 = frag_a(a_, 32, (mP + 0) * 16 + r16, khalf, 0);
    v16h af1 = frag_a(a_, 32, (mP + 1) * 16 + r16, khalf, 0);
    v16h bf0 = frag_b(b_, 32, (nP + 0) * 16 + r16, khalf, 0);
    v16h bf1 = frag_b(b_, 32, (nP + 1) * 16 + r16, khalf, 0);
    acc[0][0] = wmma16(af0, bf0, acc[0][0]);
    acc[0][1] = wmma16(af0, bf1, acc[0][1]);
    acc[1][0] = wmma16(af1, bf0, acc[1][0]);
    acc[1][1] = wmma16(af1, bf1, acc[1][1]);
    cur ^= 1;
  }

#pragma unroll
  for (int im = 0; im < 2; ++im) {
#pragma unroll
    for (int in = 0; in < 2; ++in) {
      const int n = nBlk + (nP + in) * 16 + r16;
      const float bn = bias[n];
#pragma unroll
      for (int j = 0; j < 8; ++j) {
        int m = mBlk + (mP + im) * 16 + khalf * 8 + j;
        float v = acc[im][in][j] + bn;
        if (RESID) v += resid[(size_t)m * ldo + n];
        v = apply_act<ACT>(v);
        if (OUT16)
          ((_Float16*)Out)[(size_t)m * ldo + n] = (_Float16)v;
        else
          ((float*)Out)[(size_t)m * ldo + n] = v;
      }
    }
  }
}

// ---------------------------------------------------------------------------
// Relative-position piecewise index (matches reference; RNE rounding)
// ---------------------------------------------------------------------------
__device__ __forceinline__ int pw_index(float rp) {
  float a   = fabsf(rp);
  float sgn = (rp > 0.f) ? 1.f : ((rp < 0.f) ? -1.f : 0.f);
  if (a <= 3.8f) return (int)sgn;  // 2*alpha
  float y = sgn * fminf(rintf(logf(fmaxf(a, 1e-6f) * (1.0f / 1.9f)) *
                              (3.8f / logf(6.0f))), 7.0f);
  return (int)y;
}

// ---------------------------------------------------------------------------
// Fused window attention: one block per (window, head). 49 tokens padded to 64.
// ---------------------------------------------------------------------------
template <bool SHUFFLE, bool BIAS>
__global__ __launch_bounds__(256) void attn_kernel(
    const _Float16* __restrict__ qkv,        // [NTOK][1536]
    const float* __restrict__ data,          // stride 1026, coords at +0/+1
    const float* __restrict__ bias_table,    // [15][NHEAD]
    _Float16* __restrict__ outb) {           // [NTOK][512]
  const int win  = blockIdx.x;
  const int head = blockIdx.y;
  const int tid  = threadIdx.x;
  const int lane = tid & 31, wave = tid >> 5;
  const int khalf = lane >> 4, r16 = lane & 15;

  __shared__ _Float16 sQ[64 * 64];   // [i][d]
  __shared__ _Float16 sKT[64 * 64];  // [j][d]
  __shared__ _Float16 sVT[64 * 64];  // [d][j]
  __shared__ float    sS[64 * 64];
  __shared__ _Float16 sP[64 * 64];
  __shared__ float sCx[64], sCy[64];

  {  // Q/K: one 8-half chunk per thread (64 rows x 4 chunks) -> async copies
    int p = tid >> 2, c = (tid & 3) * 8;
    if (p < WSZ) {
      size_t tok = SHUFFLE ? (size_t)p * NWIN + win : (size_t)win * WSZ + p;
      const _Float16* row = qkv + tok * 1536 + head * 64 + c;
      async_copy_b128(&sQ[p * 64 + c], row);
      async_copy_b128(&sKT[p * 64 + c], row + 512);
    } else {
      uint4 z = make_uint4(0, 0, 0, 0);
      *(uint4*)&sQ[p * 64 + c]  = z;
      *(uint4*)&sKT[p * 64 + c] = z;
    }
  }
  {  // V transposed: load 8 halves along d, scatter into [d][j]
    int j = tid >> 2, c = (tid & 3) * 8;
    union { uint4 u; _Float16 h[8]; } t;
    t.u = make_uint4(0, 0, 0, 0);
    if (j < WSZ) {
      size_t tok = SHUFFLE ? (size_t)j * NWIN + win : (size_t)win * WSZ + j;
      t.u = *(const uint4*)(qkv + tok * 1536 + 1024 + head * 64 + c);
    }
#pragma unroll
    for (int e = 0; e < 8; ++e) sVT[(c + e) * 64 + j] = t.h[e];
  }
  if (BIAS && tid < 64) {
    float cx = 0.f, cy = 0.f;
    if (tid < WSZ) {
      size_t tok = (size_t)win * WSZ + tid;
      cx = data[tok * 1026 + 0];
      cy = data[tok * 1026 + 1];
    }
    sCx[tid] = cx;
    sCy[tid] = cy;
  }
  wait_async0();
  __syncthreads();

  const int mSub  = wave & 3;
  const int nSub0 = (wave >> 2) * 2;

  {  // S = Q @ K^T
    v8f a0 = {}, a1 = {};
#pragma unroll
    for (int k0 = 0; k0 < 64; k0 += 32) {
      v16h a  = frag_a(sQ, 64, mSub * 16 + r16, khalf, k0);
      v16h b0 = frag_b(sKT, 64, nSub0 * 16 + r16, khalf, k0);
      v16h b1 = frag_b(sKT, 64, (nSub0 + 1) * 16 + r16, khalf, k0);
      a0 = wmma16(a, b0, a0);
      a1 = wmma16(a, b1, a1);
    }
#pragma unroll
    for (int j = 0; j < 8; ++j) {
      int m = mSub * 16 + khalf * 8 + j;
      sS[m * 64 + nSub0 * 16 + r16]       = a0[j] * QKSCALE;
      sS[m * 64 + (nSub0 + 1) * 16 + r16] = a1[j] * QKSCALE;
    }
  }
  __syncthreads();

  if (tid < 64) {  // softmax rows (+ relative-position bias)
    int i = tid;
    if (i < WSZ) {
      if (BIAS) {
        for (int j = 0; j < WSZ; ++j) {
          int ix = pw_index(sCx[i] - sCx[j]);
          int iy = pw_index(sCy[i] - sCy[j]);
          int idx = (ix < 0 ? -ix : ix) + (iy < 0 ? -iy : iy);
          sS[i * 64 + j] += bias_table[idx * NHEAD + head];
        }
      }
      float mx = -1e30f;
      for (int j = 0; j < WSZ; ++j) mx = fmaxf(mx, sS[i * 64 + j]);
      float sum = 0.f;
      for (int j = 0; j < WSZ; ++j) {
        float e = expf(sS[i * 64 + j] - mx);
        sum += e;
        sS[i * 64 + j] = e;
      }
      float inv = 1.0f / sum;
      for (int j = 0; j < WSZ; ++j) sP[i * 64 + j] = (_Float16)(sS[i * 64 + j] * inv);
      for (int j = WSZ; j < 64; ++j) sP[i * 64 + j] = (_Float16)0.f;
    } else {
      for (int j = 0; j < 64; ++j) sP[i * 64 + j] = (_Float16)0.f;
    }
  }
  __syncthreads();

  {  // O = P @ V
    v8f a0 = {}, a1 = {};
#pragma unroll
    for (int k0 = 0; k0 < 64; k0 += 32) {
      v16h a  = frag_a(sP, 64, mSub * 16 + r16, khalf, k0);
      v16h b0 = frag_b(sVT, 64, nSub0 * 16 + r16, khalf, k0);
      v16h b1 = frag_b(sVT, 64, (nSub0 + 1) * 16 + r16, khalf, k0);
      a0 = wmma16(a, b0, a0);
      a1 = wmma16(a, b1, a1);
    }
#pragma unroll
    for (int j = 0; j < 8; ++j) {
      int m = mSub * 16 + khalf * 8 + j;
      if (m < WSZ) {
        size_t tok = SHUFFLE ? (size_t)m * NWIN + win : (size_t)win * WSZ + m;
        outb[tok * 512 + head * 64 + nSub0 * 16 + r16]       = (_Float16)a0[j];
        outb[tok * 512 + head * 64 + (nSub0 + 1) * 16 + r16] = (_Float16)a1[j];
      }
    }
  }
}

// ---------------------------------------------------------------------------
// LayerNorm over 512 dims, one wave32 per token, 16 contiguous elems per lane.
// ---------------------------------------------------------------------------
__global__ __launch_bounds__(256) void ln_kernel(
    const float* __restrict__ x, const float* __restrict__ g,
    const float* __restrict__ b, _Float16* __restrict__ o16,
    float* __restrict__ o32, int M) {
  int gid = blockIdx.x * 256 + threadIdx.x;
  int row = gid >> 5, lane = gid & 31;
  if (row >= M) return;
  const float4* xr = (const float4*)(x + (size_t)row * DIM + lane * 16);
  float4 c0 = xr[0], c1 = xr[1], c2 = xr[2], c3 = xr[3];
  float v[16] = {c0.x, c0.y, c0.z, c0.w, c1.x, c1.y, c1.z, c1.w,
                 c2.x, c2.y, c2.z, c2.w, c3.x, c3.y, c3.z, c3.w};
  float s = 0.f;
#pragma unroll
  for (int t = 0; t < 16; ++t) s += v[t];
#pragma unroll
  for (int off = 16; off > 0; off >>= 1) s += __shfl_xor(s, off, 32);
  float mean = s * (1.0f / DIM);
  float var = 0.f;
#pragma unroll
  for (int t = 0; t < 16; ++t) { float d = v[t] - mean; var += d * d; }
#pragma unroll
  for (int off = 16; off > 0; off >>= 1) var += __shfl_xor(var, off, 32);
  var *= (1.0f / DIM);
  float inv = rsqrtf(var + 1e-5f);
  const float4* g4 = (const float4*)(g + lane * 16);
  const float4* b4 = (const float4*)(b + lane * 16);
  union { _Float16 h[16]; uint4 u[2]; } oh;
  float y[16];
#pragma unroll
  for (int q = 0; q < 4; ++q) {
    float4 gg = g4[q], bb = b4[q];
    y[q * 4 + 0] = (v[q * 4 + 0] - mean) * inv * gg.x + bb.x;
    y[q * 4 + 1] = (v[q * 4 + 1] - mean) * inv * gg.y + bb.y;
    y[q * 4 + 2] = (v[q * 4 + 2] - mean) * inv * gg.z + bb.z;
    y[q * 4 + 3] = (v[q * 4 + 3] - mean) * inv * gg.w + bb.w;
  }
#pragma unroll
  for (int t = 0; t < 16; ++t) oh.h[t] = (_Float16)y[t];
  uint4* o = (uint4*)(o16 + (size_t)row * DIM + lane * 16);
  o[0] = oh.u[0];
  o[1] = oh.u[1];
  if (o32) {
    float4* of = (float4*)(o32 + (size_t)row * DIM + lane * 16);
    of[0] = make_float4(y[0], y[1], y[2], y[3]);
    of[1] = make_float4(y[4], y[5], y[6], y[7]);
    of[2] = make_float4(y[8], y[9], y[10], y[11]);
    of[3] = make_float4(y[12], y[13], y[14], y[15]);
  }
}

__global__ __launch_bounds__(256) void gelu_kernel(float4* __restrict__ x, long n4) {
  long i = (long)blockIdx.x * blockDim.x + threadIdx.x;
  if (i < n4) {
    float4 v = x[i];
    v.x = 0.5f * v.x * (1.0f + erff(v.x * 0.7071067811865476f));
    v.y = 0.5f * v.y * (1.0f + erff(v.y * 0.7071067811865476f));
    v.z = 0.5f * v.z * (1.0f + erff(v.z * 0.7071067811865476f));
    v.w = 0.5f * v.w * (1.0f + erff(v.w * 0.7071067811865476f));
    x[i] = v;
  }
}

// scores[i] = dot(t[i,:256], ap_w2) + ap_b2 ; one wave per row, 8 elems/lane
__global__ __launch_bounds__(256) void score_kernel(
    const _Float16* __restrict__ t, const float* __restrict__ w2,
    const float* __restrict__ b2, float* __restrict__ s, int M) {
  int gid = blockIdx.x * 256 + threadIdx.x;
  int row = gid >> 5, lane = gid & 31;
  if (row >= M) return;
  union { uint4 u; _Float16 h[8]; } tv;
  tv.u = *(const uint4*)(t + (size_t)row * 256 + lane * 8);
  const float4* w4 = (const float4*)(w2 + lane * 8);
  float4 wa = w4[0], wb = w4[1];
  float acc = (float)tv.h[0] * wa.x + (float)tv.h[1] * wa.y +
              (float)tv.h[2] * wa.z + (float)tv.h[3] * wa.w +
              (float)tv.h[4] * wb.x + (float)tv.h[5] * wb.y +
              (float)tv.h[6] * wb.z + (float)tv.h[7] * wb.w;
#pragma unroll
  for (int off = 16; off > 0; off >>= 1) acc += __shfl_xor(acc, off, 32);
  if (lane == 0) s[row] = acc + b2[0];
}

// in-place softmax over n elements (single block, deterministic)
__global__ __launch_bounds__(256) void softmax_weights_kernel(float* __restrict__ s, int n) {
  __shared__ float red[256];
  int tid = threadIdx.x;
  float mx = -3.4e38f;
  for (int i = tid; i < n; i += 256) mx = fmaxf(mx, s[i]);
  red[tid] = mx;
  __syncthreads();
  for (int off = 128; off > 0; off >>= 1) {
    if (tid < off) red[tid] = fmaxf(red[tid], red[tid + off]);
    __syncthreads();
  }
  mx = red[0];
  __syncthreads();
  float sum = 0.f;
  for (int i = tid; i < n; i += 256) sum += expf(s[i] - mx);
  red[tid] = sum;
  __syncthreads();
  for (int off = 128; off > 0; off >>= 1) {
    if (tid < off) red[tid] += red[tid + off];
    __syncthreads();
  }
  float inv = 1.0f / red[0];
  for (int i = tid; i < n; i += 256) s[i] = expf(s[i] - mx) * inv;
}

// partial[b][d] = sum over 256 tokens of w[tok]*f[tok][d]
__global__ __launch_bounds__(256) void pooled_partial_kernel(
    const float* __restrict__ w, const float* __restrict__ f,
    float* __restrict__ partial) {
  int b = blockIdx.x, tid = threadIdx.x;
  int base = b * 256;
  float a0 = 0.f, a1 = 0.f;
  for (int t = 0; t < 256; ++t) {
    float wt = w[base + t];
    const float* row = f + (size_t)(base + t) * 512;
    a0 += wt * row[tid];
    a1 += wt * row[tid + 256];
  }
  partial[(size_t)b * 512 + tid]       = a0;
  partial[(size_t)b * 512 + tid + 256] = a1;
}

__global__ __launch_bounds__(256) void pooled_reduce_kernel(
    const float* __restrict__ partial, float* __restrict__ pooled, int nb) {
  int d = blockIdx.x * 256 + threadIdx.x;
  if (d < 512) {
    float s = 0.f;
    for (int b = 0; b < nb; ++b) s += partial[(size_t)b * 512 + d];
    pooled[d] = s;
  }
}

// logits -> hazards, cumprod survival, argmax
__global__ void head_kernel(const float* __restrict__ pooled,
                            const float* __restrict__ w,
                            const float* __restrict__ bias,
                            float* __restrict__ out) {
  if (threadIdx.x == 0 && blockIdx.x == 0) {
    float logits[4];
    for (int c = 0; c < 4; ++c) {
      float s = bias[c];
      for (int d = 0; d < 512; ++d) s += pooled[d] * w[d * 4 + c];
      logits[c] = s;
    }
    float S = 1.f, best = logits[0];
    int arg = 0;
    for (int c = 0; c < 4; ++c) {
      float h = 1.0f / (1.0f + expf(-logits[c]));
      out[c] = h;
      S *= (1.0f - h);
      out[4 + c] = S;
      if (c > 0 && logits[c] > best) { best = logits[c]; arg = c; }
    }
    out[8] = (float)arg;
  }
}

// ---------------------------------------------------------------------------
// Host orchestration
// ---------------------------------------------------------------------------
extern "C" void kernel_launch(void* const* d_in, const int* in_sizes, int n_in,
                              void* d_out, int out_size, void* d_ws, size_t ws_size,
                              hipStream_t stream) {
  (void)in_sizes; (void)n_in; (void)out_size; (void)ws_size;
  const float* data      = (const float*)d_in[0];
  const float* fc1_w     = (const float*)d_in[1];
  const float* fc1_b     = (const float*)d_in[2];
  const float* ln1_g     = (const float*)d_in[3];
  const float* ln1_b     = (const float*)d_in[4];
  const float* wa_qkv_w  = (const float*)d_in[5];
  const float* wa_qkv_b  = (const float*)d_in[6];
  const float* wa_btab   = (const float*)d_in[7];
  const float* wa_proj_w = (const float*)d_in[8];
  const float* wa_proj_b = (const float*)d_in[9];
  const float* n1_g      = (const float*)d_in[10];
  const float* n1_b      = (const float*)d_in[11];
  const float* sa_qkv_w  = (const float*)d_in[12];
  const float* sa_qkv_b  = (const float*)d_in[13];
  const float* sa_proj_w = (const float*)d_in[14];
  const float* sa_proj_b = (const float*)d_in[15];
  const float* n2_g      = (const float*)d_in[16];
  const float* n2_b      = (const float*)d_in[17];
  const float* mlp_w1    = (const float*)d_in[18];
  const float* mlp_b1    = (const float*)d_in[19];
  const float* mlp_w2    = (const float*)d_in[20];
  const float* mlp_b2    = (const float*)d_in[21];
  const float* n3_g      = (const float*)d_in[22];
  const float* n3_b      = (const float*)d_in[23];
  const float* ap_w1     = (const float*)d_in[24];
  const float* ap_b1     = (const float*)d_in[25];
  const float* ap_w2     = (const float*)d_in[26];
  const float* ap_b2     = (const float*)d_in[27];
  const float* fc2_w     = (const float*)d_in[28];
  const float* fc2_b     = (const float*)d_in[29];

  char* ws = (char*)d_ws;
  size_t off = 0;
  auto alloc = [&](size_t bytes) -> void* {
    void* p = ws + off;
    off += (bytes + 255) & ~(size_t)255;
    return p;
  };
  float*    buf_h   = (float*)alloc((size_t)NTOK * 512 * 4);
  _Float16* buf_ln  = (_Float16*)alloc((size_t)NTOK * 512 * 2);
  _Float16* buf_qkv = (_Float16*)alloc((size_t)NTOK * 1536 * 2);
  _Float16* buf_t   = (_Float16*)alloc((size_t)NTOK * 512 * 2);
  float*    scores  = (float*)alloc((size_t)NTOK * 4);
  float*    partial = (float*)alloc((size_t)(NTOK / 256) * 512 * 4);
  float*    pooled  = (float*)alloc(512 * 4);
  _Float16* wt_fc1  = (_Float16*)alloc((size_t)1024 * 512 * 2);
  _Float16* wt_waq  = (_Float16*)alloc((size_t)512 * 1536 * 2);
  _Float16* wt_wap  = (_Float16*)alloc((size_t)512 * 512 * 2);
  _Float16* wt_saq  = (_Float16*)alloc((size_t)512 * 1536 * 2);
  _Float16* wt_sap  = (_Float16*)alloc((size_t)512 * 512 * 2);
  _Float16* wt_m1   = (_Float16*)alloc((size_t)512 * 512 * 2);
  _Float16* wt_m2   = (_Float16*)alloc((size_t)512 * 512 * 2);
  _Float16* wt_ap1  = (_Float16*)alloc((size_t)512 * 256 * 2);

  const int M = NTOK;
  dim3 blk(256);
  dim3 lnGrid(M * 32 / 256);

  // 0. pre-transpose + convert all GEMM weights to f16 [N][K]
  wt_kernel<<<dim3((1024 * 512 + 255) / 256), blk, 0, stream>>>(fc1_w, wt_fc1, 1024, 512);
  wt_kernel<<<dim3((512 * 1536 + 255) / 256), blk, 0, stream>>>(wa_qkv_w, wt_waq, 512, 1536);
  wt_kernel<<<dim3((512 * 512 + 255) / 256), blk, 0, stream>>>(wa_proj_w, wt_wap, 512, 512);
  wt_kernel<<<dim3((512 * 1536 + 255) / 256), blk, 0, stream>>>(sa_qkv_w, wt_saq, 512, 1536);
  wt_kernel<<<dim3((512 * 512 + 255) / 256), blk, 0, stream>>>(sa_proj_w, wt_sap, 512, 512);
  wt_kernel<<<dim3((512 * 512 + 255) / 256), blk, 0, stream>>>(mlp_w1, wt_m1, 512, 512);
  wt_kernel<<<dim3((512 * 512 + 255) / 256), blk, 0, stream>>>(mlp_w2, wt_m2, 512, 512);
  wt_kernel<<<dim3((512 * 256 + 255) / 256), blk, 0, stream>>>(ap_w1, wt_ap1, 512, 256);

  // 1. h = relu(feats @ fc1_w + fc1_b)  (feats = data[:,2:], stride 1026)
  gemm_kernel<true, 1, false, false><<<dim3(512 / 64, M / 128), blk, 0, stream>>>(
      (const void*)(data + 2), 1026, wt_fc1, fc1_b, nullptr,
      (void*)buf_h, 512, M, 512, 1024);
  // 2. ln1
  ln_kernel<<<lnGrid, blk, 0, stream>>>(buf_h, ln1_g, ln1_b, buf_ln, nullptr, M);
  // 3. window-attention QKV
  gemm_kernel<false, 0, false, true><<<dim3(1536 / 64, M / 128), blk, 0, stream>>>(
      (const void*)buf_ln, 512, wt_waq, wa_qkv_b, nullptr,
      (void*)buf_qkv, 1536, M, 1536, 512);
  // 4. window attention (with relative-position bias)
  attn_kernel<false, true><<<dim3(NWIN, NHEAD), blk, 0, stream>>>(
      buf_qkv, data, wa_btab, buf_t);
  // 5. proj + residual
  gemm_kernel<false, 0, true, false><<<dim3(512 / 64, M / 128), blk, 0, stream>>>(
      (const void*)buf_t, 512, wt_wap, wa_proj_b, buf_h,
      (void*)buf_h, 512, M, 512, 512);
  // 6. f = gelu(f)
  gelu_kernel<<<dim3((M * 512 / 4) / 256), blk, 0, stream>>>((float4*)buf_h,
                                                             (long)M * 512 / 4);
  // 7. norm1
  ln_kernel<<<lnGrid, blk, 0, stream>>>(buf_h, n1_g, n1_b, buf_ln, nullptr, M);
  // 8. shuffle-attention QKV
  gemm_kernel<false, 0, false, true><<<dim3(1536 / 64, M / 128), blk, 0, stream>>>(
      (const void*)buf_ln, 512, wt_saq, sa_qkv_b, nullptr,
      (void*)buf_qkv, 1536, M, 1536, 512);
  // 9. shuffle window attention
  attn_kernel<true, false><<<dim3(NWIN, NHEAD), blk, 0, stream>>>(
      buf_qkv, nullptr, nullptr, buf_t);
  // 10. proj + residual
  gemm_kernel<false, 0, true, false><<<dim3(512 / 64, M / 128), blk, 0, stream>>>(
      (const void*)buf_t, 512, wt_sap, sa_proj_b, buf_h,
      (void*)buf_h, 512, M, 512, 512);
  // 11. norm2
  ln_kernel<<<lnGrid, blk, 0, stream>>>(buf_h, n2_g, n2_b, buf_ln, nullptr, M);
  // 12. mlp1 + gelu
  gemm_kernel<false, 2, false, true><<<dim3(512 / 64, M / 128), blk, 0, stream>>>(
      (const void*)buf_ln, 512, wt_m1, mlp_b1, nullptr,
      (void*)buf_t, 512, M, 512, 512);
  // 13. mlp2 + residual
  gemm_kernel<false, 0, true, false><<<dim3(512 / 64, M / 128), blk, 0, stream>>>(
      (const void*)buf_t, 512, wt_m2, mlp_b2, buf_h,
      (void*)buf_h, 512, M, 512, 512);
  // 14. norm3 (in-place f32 + f16 copy)
  ln_kernel<<<lnGrid, blk, 0, stream>>>(buf_h, n3_g, n3_b, buf_ln, buf_h, M);
  // 15. attention-pool hidden: tanh(f @ ap_w1 + ap_b1)
  gemm_kernel<false, 3, false, true><<<dim3(256 / 64, M / 128), blk, 0, stream>>>(
      (const void*)buf_ln, 512, wt_ap1, ap_b1, nullptr,
      (void*)buf_t, 256, M, 256, 512);
  // 16. scores
  score_kernel<<<lnGrid, blk, 0, stream>>>(buf_t, ap_w2, ap_b2, scores, M);
  // 17. softmax over tokens (in place)
  softmax_weights_kernel<<<dim3(1), blk, 0, stream>>>(scores, M);
  // 18/19. pooled = weights @ feat (deterministic two-phase)
  pooled_partial_kernel<<<dim3(M / 256), blk, 0, stream>>>(scores, buf_h, partial);
  pooled_reduce_kernel<<<dim3(2), blk, 0, stream>>>(partial, pooled, M / 256);
  // 20. logits -> hazards, survival, argmax
  head_kernel<<<dim3(1), dim3(64), 0, stream>>>(pooled, fc2_w, fc2_b, (float*)d_out);
}
